// NCT_14104672600386
// MI455X (gfx1250) — matrix operand order repaired and checked by reading, not compiled, not run on hardware
//
#include <hip/hip_runtime.h>
#include <cmath>

typedef __attribute__((ext_vector_type(16))) _Float16 v16h;
typedef __attribute__((ext_vector_type(8)))  _Float16 v8h;
typedef __attribute__((ext_vector_type(8)))  float    v8f;

#define B_ROWS   131072
#define D_INP    256
#define HID      128
#define E_DIM    128
#define K_CB     9
#define NCB_ELT  (K_CB*E_DIM)   // 1152
#define NSLOPE   0.01f

#define WAVES_PER_WG 8
#define THREADS      256
#define ROWS_PER_WG  128

// ---- LDS layout -----------------------------------------------------------
// wbuf (f16): encoder phase: We1[128x256]@0, We2@32768, We3@49152, We4@65536
//             decoder phase: Wd1[128x256]@0, Wd2@32768, Wd3@49152
#define WBUF_HALFS   81920
// per-wave activation scratch: ping[16x128] + pong[16x128]
#define SCR_HALFS_PW (16*256)
#define SCR_HALFS    (WAVES_PER_WG*SCR_HALFS_PW)
// float region (indices in floats)
#define F_BIAS     0                       // 4 x 128 (per phase)
#define F_BD4      512
#define F_LOSSVQ   513
#define F_LOSSMEAN 514
#define F_CBNORM   516                     // 9
#define F_CB       528                     // 9*128
#define F_MEANCB   (F_CB + K_CB*E_DIM)     // 1680
#define F_TOTAL    (F_MEANCB + E_DIM)      // 1808

#define SMEM_BYTES ((size_t)(WBUF_HALFS + SCR_HALFS)*2 + (size_t)F_TOTAL*4)

// ---- Tensor Data Mover availability ---------------------------------------
#if defined(__AMDGCN__) && __has_builtin(__builtin_amdgcn_tensor_load_to_lds)
#define HAVE_TDM 1
typedef unsigned int tdm_u32x4 __attribute__((ext_vector_type(4)));
typedef int          tdm_i32x4 __attribute__((ext_vector_type(4)));
typedef int          tdm_i32x8 __attribute__((ext_vector_type(8)));
#endif

// ---- helpers --------------------------------------------------------------
__device__ __forceinline__ void load_w16(const float* __restrict__ g, _Float16* s, int n, int tid) {
  for (int i = tid; i < n; i += THREADS) s[i] = (_Float16)g[i];
}
__device__ __forceinline__ void load_f32s(const float* __restrict__ g, float* s, int n, int tid) {
  for (int i = tid; i < n; i += THREADS) s[i] = g[i];
}

// A-fragment (16x32 f16) from LDS row-major [16][128] activations.
// ISA layout: lane<16 holds K {kb..kb+7, kb+16..kb+23}; lane>=16 the +8 runs.
__device__ __forceinline__ v16h lds_frag(const _Float16* s, int kb, int l16, int hi) {
  const _Float16* p = s + l16*HID + kb + hi*8;
  union { v16h v; v8h h[2]; } u;
  u.h[0] = *(const v8h*)p;
  u.h[1] = *(const v8h*)(p + 16);
  return u.v;
}

// One 16-row x 128-col layer: Out = act(A[16xKD] @ W^T + b).
// W is f16 LDS, original [N=128][KD] row-major => B-fragment [k][n] = W[n][k]:
// lane reads halfs {kb+8*hi..+7, kb+16+8*hi..+7} of row (16n + l16).
template <int KD, typename AF>
__device__ __forceinline__ void wlayer(AF aload, const _Float16* __restrict__ W,
                                       const float* __restrict__ bias,
                                       _Float16* __restrict__ Out,
                                       bool act, int l16, int hi) {
  v8f acc[8] = {};
  for (int kb = 0; kb < KD; kb += 32) {
    const v16h a = aload(kb);
#pragma unroll
    for (int n = 0; n < 8; ++n) {
      const _Float16* wp = W + (n*16 + l16)*KD + kb + hi*8;
      union { v16h v; v8h h[2]; } ub;
      ub.h[0] = *(const v8h*)wp;
      ub.h[1] = *(const v8h*)(wp + 16);
      const v16h b = ub.v;
      acc[n] = __builtin_amdgcn_wmma_f32_16x16x32_f16(
          false, a, false, b, (short)0, acc[n], false, false);
    }
  }
#pragma unroll
  for (int n = 0; n < 8; ++n) {
    const int N = n*16 + l16;
    const float bv = bias[N];
#pragma unroll
    for (int d = 0; d < 8; ++d) {
      float v = acc[n][d] + bv;                 // C/D: M = d + 8*hi, N = 16n + l16
      if (act) v = (v > 0.f) ? v : NSLOPE * v;  // leaky relu
      Out[(d + hi*8)*HID + N] = (_Float16)v;
    }
  }
}

// Encoder chain for one wave's 16 rows; z_e (fp->f16) ends in `pong`.
__device__ __forceinline__ void encode_rows(const float* __restrict__ xr,
                                            _Float16* ping, _Float16* pong,
                                            const _Float16* wbuf, const float* fbuf,
                                            int l16, int hi) {
  auto aglb = [&](int kb) -> v16h {   // A-frag straight from global fp32 input
    const float4* q0 = (const float4*)(xr + kb + hi*8);
    const float4* q1 = (const float4*)(xr + kb + 16 + hi*8);
    float4 a0 = q0[0], a1 = q0[1], b0 = q1[0], b1 = q1[1];
    union { v16h v; _Float16 e[16]; } u;
    u.e[0]=(_Float16)a0.x;  u.e[1]=(_Float16)a0.y;  u.e[2]=(_Float16)a0.z;  u.e[3]=(_Float16)a0.w;
    u.e[4]=(_Float16)a1.x;  u.e[5]=(_Float16)a1.y;  u.e[6]=(_Float16)a1.z;  u.e[7]=(_Float16)a1.w;
    u.e[8]=(_Float16)b0.x;  u.e[9]=(_Float16)b0.y;  u.e[10]=(_Float16)b0.z; u.e[11]=(_Float16)b0.w;
    u.e[12]=(_Float16)b1.x; u.e[13]=(_Float16)b1.y; u.e[14]=(_Float16)b1.z; u.e[15]=(_Float16)b1.w;
    return u.v;
  };
  auto aP = [&](int kb) -> v16h { return lds_frag(ping, kb, l16, hi); };
  auto aQ = [&](int kb) -> v16h { return lds_frag(pong, kb, l16, hi); };
  wlayer<D_INP>(aglb, wbuf,         fbuf + F_BIAS,       ping, true,  l16, hi);
  wlayer<HID>  (aP,   wbuf + 32768, fbuf + F_BIAS + 128, pong, true,  l16, hi);
  wlayer<HID>  (aQ,   wbuf + 49152, fbuf + F_BIAS + 256, ping, true,  l16, hi);
  wlayer<HID>  (aP,   wbuf + 65536, fbuf + F_BIAS + 384, pong, false, l16, hi);
}

// VQ for one wave's 16 rows (z_e f16 in Z). Lane pair (l16, l16+16) splits the
// 128 dims; after shfl_xor both halves hold full sums so argmin is uniform.
__device__ __forceinline__ int vq_rows(const _Float16* __restrict__ Z,
                                       const float* __restrict__ fbuf,
                                       int l16, int hi, float& lv, float& lm) {
  const float* cbf = fbuf + F_CB;
  const float* cbn = fbuf + F_CBNORM;
  const float* mcb = fbuf + F_MEANCB;
  const int off = hi*64;
  const _Float16* zr = Z + l16*E_DIM + off;
  float ss = 0.f;
  float dots[K_CB];
#pragma unroll
  for (int j = 0; j < K_CB; ++j) dots[j] = 0.f;
  for (int k = 0; k < 64; ++k) {
    const float z = (float)zr[k];
    ss += z*z;
#pragma unroll
    for (int j = 0; j < K_CB; ++j) dots[j] += z * cbf[j*E_DIM + off + k];
  }
  ss += __shfl_xor(ss, 16, 32);
#pragma unroll
  for (int j = 0; j < K_CB; ++j) dots[j] += __shfl_xor(dots[j], 16, 32);
  int best = 0;
  float bd = ss - 2.f*dots[0] + cbn[0];
#pragma unroll
  for (int j = 1; j < K_CB; ++j) {
    const float dj = ss - 2.f*dots[j] + cbn[j];
    if (dj < bd) { bd = dj; best = j; }
  }
  for (int k = 0; k < 64; ++k) {     // loss partials (each element counted once)
    const float z = (float)zr[k];
    const float q = cbf[best*E_DIM + off + k];
    const float m = mcb[off + k];
    lv += (q - z)*(q - z);
    lm += (z - m)*(z - m);
  }
  return best;
}

// Decoder: input is [codebook[i0], codebook[i1]] (straight-through forward
// value), built as A-fragments directly from the LDS codebook (no Xd buffer).
__device__ __forceinline__ float decode_rows(int i0, int i1,
                                             _Float16* ping, _Float16* pong,
                                             const _Float16* wbuf, const float* fbuf,
                                             int l16, int hi) {
  auto acb = [&](int kb) -> v16h {
    const int code = (kb & 128) ? i1 : i0;
    const float4* q0 = (const float4*)(fbuf + F_CB + code*E_DIM + (kb & 127) + hi*8);
    const float4* q1 = q0 + 4;  // +16 floats
    float4 a0 = q0[0], a1 = q0[1], b0 = q1[0], b1 = q1[1];
    union { v16h v; _Float16 e[16]; } u;
    u.e[0]=(_Float16)a0.x;  u.e[1]=(_Float16)a0.y;  u.e[2]=(_Float16)a0.z;  u.e[3]=(_Float16)a0.w;
    u.e[4]=(_Float16)a1.x;  u.e[5]=(_Float16)a1.y;  u.e[6]=(_Float16)a1.z;  u.e[7]=(_Float16)a1.w;
    u.e[8]=(_Float16)b0.x;  u.e[9]=(_Float16)b0.y;  u.e[10]=(_Float16)b0.z; u.e[11]=(_Float16)b0.w;
    u.e[12]=(_Float16)b1.x; u.e[13]=(_Float16)b1.y; u.e[14]=(_Float16)b1.z; u.e[15]=(_Float16)b1.w;
    return u.v;
  };
  auto aP = [&](int kb) -> v16h { return lds_frag(ping, kb, l16, hi); };
  auto aQ = [&](int kb) -> v16h { return lds_frag(pong, kb, l16, hi); };
  wlayer<2*E_DIM>(acb, wbuf,         fbuf + F_BIAS,       ping, true, l16, hi);
  wlayer<HID>    (aP,  wbuf + 32768, fbuf + F_BIAS + 128, pong, true, l16, hi);
  wlayer<HID>    (aQ,  wbuf + 49152, fbuf + F_BIAS + 256, ping, true, l16, hi);
  // final 128 -> 1 dot + tanh on VALU (Wd4 in fbuf[F_BIAS+384..511])
  const float* wd4 = fbuf + F_BIAS + 384;
  float s = 0.f;
  const int off = hi*64;
  for (int k = 0; k < 64; ++k) s += (float)ping[l16*HID + off + k] * wd4[off + k];
  s += __shfl_xor(s, 16, 32);
  return tanhf(s + fbuf[F_BD4]);
}

// ---- kernels --------------------------------------------------------------
__global__ void nct_zero_losses(float* __restrict__ out) {
  if (threadIdx.x == 0) { out[B_ROWS] = 0.f; out[B_ROWS + 1] = 0.f; }
}

__global__ __launch_bounds__(THREADS, 1)
void nct_vqvae_fused(const float* __restrict__ cA,  const float* __restrict__ cB,
                     const float* __restrict__ We1, const float* __restrict__ be1,
                     const float* __restrict__ We2, const float* __restrict__ be2,
                     const float* __restrict__ We3, const float* __restrict__ be3,
                     const float* __restrict__ We4, const float* __restrict__ be4,
                     const float* __restrict__ cbg,
                     const float* __restrict__ Wd1, const float* __restrict__ bd1,
                     const float* __restrict__ Wd2, const float* __restrict__ bd2,
                     const float* __restrict__ Wd3, const float* __restrict__ bd3,
                     const float* __restrict__ Wd4, const float* __restrict__ bd4,
                     float* __restrict__ out) {
  extern __shared__ __align__(64) char smem_raw[];
  _Float16* wbuf = (_Float16*)smem_raw;
  _Float16* scr  = wbuf + WBUF_HALFS;
  float*    fbuf = (float*)(smem_raw + (size_t)(WBUF_HALFS + SCR_HALFS)*2);

  const int tid  = threadIdx.x;
  const int lane = tid & 31;
  const int wave = tid >> 5;
  const int l16  = lane & 15;
  const int hi   = lane >> 4;

  // ---- phase 0: stage encoder weights (fp32 -> f16) + constants into LDS
  load_w16(We1, wbuf,          128*256, tid);
  load_w16(We2, wbuf + 32768,  128*128, tid);
  load_w16(We3, wbuf + 49152,  128*128, tid);
  load_w16(We4, wbuf + 65536,  128*128, tid);
  load_f32s(be1, fbuf + F_BIAS,       128, tid);
  load_f32s(be2, fbuf + F_BIAS + 128, 128, tid);
  load_f32s(be3, fbuf + F_BIAS + 256, 128, tid);
  load_f32s(be4, fbuf + F_BIAS + 384, 128, tid);

#ifdef HAVE_TDM
  // Codebook staged by the Tensor Data Mover: one 1-D D# (1152 fp32 elements).
  if (wave == 0) {
    const unsigned long long ga = (unsigned long long)(uintptr_t)cbg;
    const unsigned int lds_off  = (unsigned int)(uintptr_t)(fbuf + F_CB);
    tdm_u32x4 g0 = {
        1u,                                                  // count=1, user D#
        lds_off,                                             // lds_addr[63:32]
        (unsigned int)(ga & 0xffffffffu),                    // global_addr lo
        (unsigned int)((ga >> 32) & 0x01ffffffu) | (2u<<30)  // addr hi | type=2
    };
    tdm_i32x8 g1 = {
        (int)(2u << 16),                        // data_size=4B; mask=0
        (int)((unsigned)NCB_ELT << 16),         // tensor_dim0[15:0] @ bits63:48
        (int)(1u << 16),                        // tensor_dim1=1    @ bits95:80
        (int)((unsigned)NCB_ELT << 16),         // tile_dim0        @ bits127:112
        0,                                      // tile_dim1=0, tile_dim2=0
        (int)NCB_ELT,                           // tensor_dim0_stride lo32
        0, 0
    };
    tdm_i32x4 z4 = {0, 0, 0, 0};
#if __clang_major__ >= 23
    tdm_i32x8 z8 = {0, 0, 0, 0, 0, 0, 0, 0};
    __builtin_amdgcn_tensor_load_to_lds(g0, g1, z4, z4, z8, 0);
#else
    __builtin_amdgcn_tensor_load_to_lds(g0, g1, z4, z4, 0);
#endif
#if __has_builtin(__builtin_amdgcn_s_wait_tensorcnt)
    __builtin_amdgcn_s_wait_tensorcnt(0);
#endif
  }
#else
  load_f32s(cbg, fbuf + F_CB, NCB_ELT, tid);
#endif

  // warm L2 with the decoder weights while the encoder phase runs
  for (int i = tid*32; i < 128*256; i += THREADS*32) __builtin_prefetch(Wd1 + i, 0, 1);
  for (int i = tid*32; i < 128*128; i += THREADS*32) {
    __builtin_prefetch(Wd2 + i, 0, 1);
    __builtin_prefetch(Wd3 + i, 0, 1);
  }

  if (tid == 0) { fbuf[F_LOSSVQ] = 0.f; fbuf[F_LOSSMEAN] = 0.f; fbuf[F_BD4] = bd4[0]; }
  __syncthreads();
  if (tid < K_CB) {                              // ||c_j||^2
    float s = 0.f;
    for (int k = 0; k < E_DIM; ++k) { float c = fbuf[F_CB + tid*E_DIM + k]; s += c*c; }
    fbuf[F_CBNORM + tid] = s;
  }
  if (tid >= 32 && tid < 32 + E_DIM) {           // mean_cb[k]
    const int k = tid - 32;
    float m = 0.f;
    for (int j = 0; j < K_CB; ++j) m += fbuf[F_CB + j*E_DIM + k];
    fbuf[F_MEANCB + k] = m * (1.0f / (float)K_CB);
  }
  __syncthreads();

  _Float16* ping = scr + wave*SCR_HALFS_PW;
  _Float16* pong = ping + 16*HID;
  const size_t rowBase = (size_t)blockIdx.x * ROWS_PER_WG + (size_t)wave * 16;

  // prefetch this wave's comp_B rows while comp_A is encoded
  {
    const float* pfB = cB + (rowBase + (size_t)l16)*D_INP + (size_t)hi*128;
#pragma unroll
    for (int i = 0; i < 4; ++i) __builtin_prefetch(pfB + i*32, 0, 1);
  }

  float lv = 0.f, lm = 0.f;
  // ---- siamese encoder + VQ
  encode_rows(cA + (rowBase + (size_t)l16)*D_INP, ping, pong, wbuf, fbuf, l16, hi);
  const int iA = vq_rows(pong, fbuf, l16, hi, lv, lm);
  encode_rows(cB + (rowBase + (size_t)l16)*D_INP, ping, pong, wbuf, fbuf, l16, hi);
  const int iB = vq_rows(pong, fbuf, l16, hi, lv, lm);

  // ---- phase 1: overlay decoder weights over encoder weights in LDS
  __syncthreads();
  load_w16(Wd1, wbuf,          128*256, tid);
  load_w16(Wd2, wbuf + 32768,  128*128, tid);
  load_w16(Wd3, wbuf + 49152,  128*128, tid);
  load_f32s(bd1, fbuf + F_BIAS,       128, tid);
  load_f32s(bd2, fbuf + F_BIAS + 128, 128, tid);
  load_f32s(bd3, fbuf + F_BIAS + 256, 128, tid);
  load_f32s(Wd4, fbuf + F_BIAS + 384, 128, tid);
  __syncthreads();

  // ---- two decoder passes (swapped halves) -> residual
  const float x1 = decode_rows(iA, iB, ping, pong, wbuf, fbuf, l16, hi);
  const float x2 = decode_rows(iB, iA, ping, pong, wbuf, fbuf, l16, hi);

  if (hi == 0) {
    const size_t r = rowBase + (size_t)l16;
    out[r] = 0.5f * (x1 - x2);                       // residual [B]
    out[(size_t)B_ROWS + 2 + r]   = (float)iA;       // idx_A [B]
    out[(size_t)2*B_ROWS + 2 + r] = (float)iB;       // idx_B [B]
  }

  // ---- loss reduction: wave -> LDS -> one global atomic per workgroup
#pragma unroll
  for (int o = 16; o >= 1; o >>= 1) {
    lv += __shfl_xor(lv, o, 32);
    lm += __shfl_xor(lm, o, 32);
  }
  if (lane == 0) {
    atomicAdd(&fbuf[F_LOSSVQ], lv);
    atomicAdd(&fbuf[F_LOSSMEAN], lm);
  }
  __syncthreads();
  if (tid == 0) {
    const float sc = 1.0f / (2.0f * (float)B_ROWS * (float)E_DIM);
    atomicAdd(out + B_ROWS,     fbuf[F_LOSSVQ]   * sc);
    atomicAdd(out + B_ROWS + 1, fbuf[F_LOSSMEAN] * sc);
  }
}

// ---- host entry -----------------------------------------------------------
extern "C" void kernel_launch(void* const* d_in, const int* in_sizes, int n_in,
                              void* d_out, int out_size, void* d_ws, size_t ws_size,
                              hipStream_t stream) {
  (void)in_sizes; (void)n_in; (void)out_size; (void)d_ws; (void)ws_size;
  const float* cA  = (const float*)d_in[0];
  const float* cB  = (const float*)d_in[1];
  const float* We1 = (const float*)d_in[2];
  const float* be1 = (const float*)d_in[3];
  const float* We2 = (const float*)d_in[4];
  const float* be2 = (const float*)d_in[5];
  const float* We3 = (const float*)d_in[6];
  const float* be3 = (const float*)d_in[7];
  const float* We4 = (const float*)d_in[8];
  const float* be4 = (const float*)d_in[9];
  const float* cbg = (const float*)d_in[10];
  const float* Wd1 = (const float*)d_in[11];
  const float* bd1 = (const float*)d_in[12];
  const float* Wd2 = (const float*)d_in[13];
  const float* bd2 = (const float*)d_in[14];
  const float* Wd3 = (const float*)d_in[15];
  const float* bd3 = (const float*)d_in[16];
  const float* Wd4 = (const float*)d_in[17];
  const float* bd4 = (const float*)d_in[18];
  float* out = (float*)d_out;

  (void)hipFuncSetAttribute((const void*)nct_vqvae_fused,
                            hipFuncAttributeMaxDynamicSharedMemorySize,
                            (int)SMEM_BYTES);

  nct_zero_losses<<<dim3(1), dim3(32), 0, stream>>>(out);
  nct_vqvae_fused<<<dim3(B_ROWS / ROWS_PER_WG), dim3(THREADS), SMEM_BYTES, stream>>>(
      cA, cB, We1, be1, We2, be2, We3, be3, We4, be4, cbg,
      Wd1, bd1, Wd2, bd2, Wd3, bd3, Wd4, bd4, out);
}
